// GCN_83219286327606
// MI455X (gfx1250) — matrix operand order repaired and checked by reading, not compiled, not run on hardware
//
#include <hip/hip_runtime.h>
#include <hip/hip_bf16.h>

// GraphConv GNN on gfx1250 (MI455X, wave32, WMMA, LDS-staged weights).
//
// Pipeline:
//   1. prep_w: transpose+convert 512x512 f32 weights -> N-major bf16
//   2. cvt x[:N1] -> bf16 (root-term A input)
//   3. zero + scatter_add (global_atomic_add_f32) over E1 edges; cvt agg1 -> bf16
//   4. gemm_wmma: block = 8 waves sharing one 64-col n-group (weights staged in
//      LDS once, 8x less L2 weight traffic); each wave = 16x64 tile, 4 accs.
//        hb = relu(agg1b@Wrel1 + brel1 + xb@Wroot1)   [bf16 out]
//   5. zero + scatter_add from bf16 hb over E2 edges; cvt agg2 -> bf16
//   6. h2b = agg2b@Wrel2 + brel2 + hb[:N2]@Wroot2     [bf16 out]
//   7. h3  = h2b@Wlin + blin                          [f32 out]
//   8. head: 4000x512x10 FMA -> d_out
//
// Roofline: edge gather (~1GB @ 23.3TB/s) dominates; with LDS staging the GEMM
// L2 traffic drops to ~0.8GB (A) + 0.16GB (B); WMMA FLOPs are negligible.

typedef __attribute__((ext_vector_type(16))) __bf16 v16bf;
typedef __attribute__((ext_vector_type(8)))  __bf16 v8bf;
typedef __attribute__((ext_vector_type(8)))  float  v8f;

#define FDIM 512        // F_IN == H == 512; log2 = 9
#define BROW 520        // padded LDS row stride in bf16 (+16B => banks 0,4,8..60)

// ---------------------------------------------------------------- utilities

__global__ void zero_f32_kernel(float* __restrict__ p, long n4) {
    long i = (long)blockIdx.x * blockDim.x + threadIdx.x;
    if (i < n4) ((float4*)p)[i] = make_float4(0.f, 0.f, 0.f, 0.f);
}

__global__ void cvt_f32_bf16_kernel(const float* __restrict__ in,
                                    __bf16* __restrict__ out, long n4) {
    long i = (long)blockIdx.x * blockDim.x + threadIdx.x;
    if (i >= n4) return;
    const float4 v = ((const float4*)in)[i];
    __bf16 r[4] = {(__bf16)v.x, (__bf16)v.y, (__bf16)v.z, (__bf16)v.w};
    *(ushort4*)&((ushort*)out)[i * 4] = *(ushort4*)r;
}

// W[k, n] (512x512 row-major f32) -> Wt[n, k] bf16 (N-major: a 64-column slab
// of Wt is one contiguous 64KB block, ideal for the LDS staging copy).
__global__ void prep_w_kernel(const float* __restrict__ W, __bf16* __restrict__ Wt) {
    int idx = blockIdx.x * blockDim.x + threadIdx.x;   // 512*512 threads
    int k = idx >> 9, n = idx & 511;
    Wt[((size_t)n << 9) + k] = (__bf16)W[idx];
}

// ------------------------------------------------------------- scatter-add
// agg[dst[e], :] += X[src[e], :]  -- 128 threads per edge, 4 floats each.
__global__ void scatter_add_f32_kernel(const float* __restrict__ X,
                                       const int* __restrict__ src,
                                       const int* __restrict__ dst,
                                       float* __restrict__ agg, long total) {
    long gid = (long)blockIdx.x * blockDim.x + threadIdx.x;
    if (gid >= total) return;
    int e = (int)(gid >> 7);
    int c = ((int)gid & 127) << 2;
    const float4 v = *(const float4*)(X + ((size_t)src[e] << 9) + c);
    float* p = agg + ((size_t)dst[e] << 9) + c;
    atomicAdd(p + 0, v.x);
    atomicAdd(p + 1, v.y);
    atomicAdd(p + 2, v.z);
    atomicAdd(p + 3, v.w);
}

// Same but gathers from a bf16 source (halves gather bytes for layer 2).
__global__ void scatter_add_bf16_kernel(const __bf16* __restrict__ X,
                                        const int* __restrict__ src,
                                        const int* __restrict__ dst,
                                        float* __restrict__ agg, long total) {
    long gid = (long)blockIdx.x * blockDim.x + threadIdx.x;
    if (gid >= total) return;
    int e = (int)(gid >> 7);
    int c = ((int)gid & 127) << 2;
    const __bf16* s = X + ((size_t)src[e] << 9) + c;
    ushort4 raw = *(const ushort4*)s;            // 4 bf16 = 8B
    const __bf16* bv = (const __bf16*)&raw;
    float* p = agg + ((size_t)dst[e] << 9) + c;
    atomicAdd(p + 0, (float)bv[0]);
    atomicAdd(p + 1, (float)bv[1]);
    atomicAdd(p + 2, (float)bv[2]);
    atomicAdd(p + 3, (float)bv[3]);
}

// ------------------------------------------------------------- WMMA GEMM
// A-fragment (16x32 bf16, MxK) per ISA 7.12.2, source row-major bf16 [M,512]:
//   lanes 0-15:  M=lane,    elems 0-7 -> K=k0..k0+7,    elems 8-15 -> K=k0+16..+23
//   lanes 16-31: M=lane-16, elems 0-7 -> K=k0+8..k0+15, elems 8-15 -> K=k0+24..+31
__device__ __forceinline__ v16bf a_frag(const __bf16* __restrict__ A, int mBase,
                                        int k0, int lane) {
    const __bf16* row = A + ((size_t)(mBase + (lane & 15)) << 9) + k0 + ((lane >= 16) ? 8 : 0);
    v8bf lo = *(const v8bf*)row;
    v8bf hi = *(const v8bf*)(row + 16);
    v16bf r;
#pragma unroll
    for (int e = 0; e < 8; ++e) { r[e] = lo[e]; r[8 + e] = hi[e]; }
    return r;
}

// B-fragment (32x16 bf16, KxN) read from the LDS-staged N-major slab:
// lanes 0-15 hold K=k0..k0+7,k0+16..+23 at local col (16t + lane), lanes 16-31
// K=k0+8..+15,k0+24..+31. Two 16B ds loads (row stride 1040B is 16B aligned).
__device__ __forceinline__ v16bf b_frag_lds(const __bf16* Bsh, int t, int k0, int lane) {
    const __bf16* p = Bsh + (size_t)(t * 16 + (lane & 15)) * BROW + k0 + ((lane >= 16) ? 8 : 0);
    v8bf lo = *(const v8bf*)p;
    v8bf hi = *(const v8bf*)(p + 16);
    v16bf r;
#pragma unroll
    for (int e = 0; e < 8; ++e) { r[e] = lo[e]; r[8 + e] = hi[e]; }
    return r;
}

// Block = 256 threads = 8 waves, all sharing ONE 64-column n-group whose full
// K=512 weight slab(s) are staged in LDS once (64KB per matrix, padded rows).
// Each wave computes a 16(M) x 64(N) tile with 4 independent accumulators.
// gridDim.x = 8 * ceil(mTiles/8);  blockIdx: nGroup = blk & 7, mGroup = blk >> 3.
template <bool RELU, bool DUAL, bool OUT_BF16>
__global__ void __launch_bounds__(256)
gemm_wmma_kernel(const __bf16* __restrict__ A1, const __bf16* __restrict__ W1t,
                 const __bf16* __restrict__ A2, const __bf16* __restrict__ W2t,
                 const float* __restrict__ bias, void* __restrict__ out,
                 int mTiles) {
    constexpr int NB = DUAL ? 2 : 1;
    __shared__ __bf16 Bsh[NB * 64 * BROW];

    const int nGroup = blockIdx.x & 7;
    const int mGroup = blockIdx.x >> 3;
    const int nBase  = nGroup << 6;
    const int lane   = threadIdx.x & 31;
    const int mTile  = mGroup * 8 + ((int)threadIdx.x >> 5);

    // Cooperative stage: 64 contiguous N-major rows (64KB) per matrix -> LDS
    // with +16B/row padding. 4096 float4 moves per matrix across 256 threads.
#pragma unroll
    for (int m = 0; m < NB; ++m) {
        const float4* src = (const float4*)((m == 0 ? W1t : W2t) + ((size_t)nBase << 9));
        __bf16* dstBase = Bsh + (size_t)m * 64 * BROW;
#pragma unroll
        for (int it = 0; it < 16; ++it) {
            int i = it * 256 + (int)threadIdx.x;      // 0..4095
            int row = i >> 6, col = i & 63;           // col in 16B units
            *(float4*)(dstBase + (size_t)row * BROW + col * 8) = src[i];
        }
    }
    __syncthreads();

    if (mTile >= mTiles) return;
    const int mBase = mTile << 4;

    v8f acc[4] = {{}, {}, {}, {}};
#pragma unroll 2
    for (int k0 = 0; k0 < FDIM; k0 += 32) {
        v16bf a1 = a_frag(A1, mBase, k0, lane);
        v16bf a2;
        if (DUAL) a2 = a_frag(A2, mBase, k0, lane);
#pragma unroll
        for (int t = 0; t < 4; ++t) {
            v16bf b1 = b_frag_lds(Bsh, t, k0, lane);
            acc[t] = __builtin_amdgcn_wmma_f32_16x16x32_bf16(false, a1, false, b1,
                                                             (short)0, acc[t], false, false);
            if (DUAL) {
                v16bf b2 = b_frag_lds(Bsh + 64 * BROW, t, k0, lane);
                acc[t] = __builtin_amdgcn_wmma_f32_16x16x32_bf16(false, a2, false, b2,
                                                                 (short)0, acc[t], false, false);
            }
        }
    }

    // C/D layout: VGPR r -> row mBase+(hi?8:0)+r, col nBase+16t+(lane&15).
    const int r0 = mBase + ((lane >= 16) ? 8 : 0);
#pragma unroll
    for (int t = 0; t < 4; ++t) {
        int col = nBase + 16 * t + (lane & 15);
        float bv = bias[col];
#pragma unroll
        for (int r = 0; r < 8; ++r) {
            float v = acc[t][r] + bv;
            if (RELU) v = fmaxf(v, 0.f);
            size_t idx = ((size_t)(r0 + r) << 9) + col;
            if (OUT_BF16) ((__bf16*)out)[idx] = (__bf16)v;
            else          ((float*)out)[idx] = v;
        }
    }
}

// ------------------------------------------------------------------- head
// out[n, c] = h3[n, :] . Whead[:, c] + bhead[c]   (C=10, trivial FLOPs)
__global__ void head_kernel(const float* __restrict__ h3, const float* __restrict__ Wh,
                            const float* __restrict__ bh, float* __restrict__ out,
                            int total) {
    int idx = blockIdx.x * blockDim.x + threadIdx.x;
    if (idx >= total) return;
    int n = idx / 10, c = idx - n * 10;
    float acc = bh[c];
    const float* row = h3 + ((size_t)n << 9);
#pragma unroll 8
    for (int k = 0; k < FDIM; ++k) acc += row[k] * Wh[k * 10 + c];
    out[idx] = acc;
}

// ---------------------------------------------------------------- launcher

extern "C" void kernel_launch(void* const* d_in, const int* in_sizes, int n_in,
                              void* d_out, int out_size, void* d_ws, size_t ws_size,
                              hipStream_t stream) {
    const float* x      = (const float*)d_in[0];
    const int*   src1   = (const int*)d_in[1];
    const int*   dst1   = (const int*)d_in[2];
    const int*   src2   = (const int*)d_in[3];
    const int*   dst2   = (const int*)d_in[4];
    const float* Wrel1  = (const float*)d_in[5];
    const float* brel1  = (const float*)d_in[6];
    const float* Wroot1 = (const float*)d_in[7];
    const float* Wrel2  = (const float*)d_in[8];
    const float* brel2  = (const float*)d_in[9];
    const float* Wroot2 = (const float*)d_in[10];
    const float* Wlin   = (const float*)d_in[11];
    const float* blin   = (const float*)d_in[12];
    const float* Whead  = (const float*)d_in[13];
    const float* bhead  = (const float*)d_in[14];
    float* out = (float*)d_out;

    const int N1 = 20000, N2 = 4000, E1 = 500000, E2 = 100000;

    // Carve scratch (256B aligned), ~130MB total.
    char* ws = (char*)d_ws;
    size_t off = 0;
    auto carve = [&](size_t bytes) -> char* {
        off = (off + 255) & ~(size_t)255;
        char* p = ws + off;
        off += bytes;
        return p;
    };
    const size_t WB  = (size_t)512 * 512 * sizeof(__bf16);
    const size_t R1B = (size_t)N1 * FDIM;     // elements per N1 x 512 matrix
    const size_t R2B = (size_t)N2 * FDIM;
    __bf16* Wt_rel1  = (__bf16*)carve(WB);
    __bf16* Wt_root1 = (__bf16*)carve(WB);
    __bf16* Wt_rel2  = (__bf16*)carve(WB);
    __bf16* Wt_root2 = (__bf16*)carve(WB);
    __bf16* Wt_lin   = (__bf16*)carve(WB);
    __bf16* xb    = (__bf16*)carve(R1B * sizeof(__bf16));   // x[:N1] in bf16
    float*  agg1  = (float*) carve(R1B * sizeof(float));
    __bf16* agg1b = (__bf16*)carve(R1B * sizeof(__bf16));
    __bf16* hb    = (__bf16*)carve(R1B * sizeof(__bf16));
    float*  agg2  = (float*) carve(R2B * sizeof(float));
    __bf16* agg2b = (__bf16*)carve(R2B * sizeof(__bf16));
    __bf16* h2b   = (__bf16*)carve(R2B * sizeof(__bf16));
    float*  h3    = (float*) carve(R2B * sizeof(float));

    // 1) Weight prep + x[:N1] -> bf16
    prep_w_kernel<<<1024, 256, 0, stream>>>(Wrel1,  Wt_rel1);
    prep_w_kernel<<<1024, 256, 0, stream>>>(Wroot1, Wt_root1);
    prep_w_kernel<<<1024, 256, 0, stream>>>(Wrel2,  Wt_rel2);
    prep_w_kernel<<<1024, 256, 0, stream>>>(Wroot2, Wt_root2);
    prep_w_kernel<<<1024, 256, 0, stream>>>(Wlin,   Wt_lin);
    long n4x = (long)R1B / 4;
    cvt_f32_bf16_kernel<<<(int)((n4x + 255) / 256), 256, 0, stream>>>(x, xb, n4x);

    // 2) Layer 1 aggregation: agg1 = segment_sum(x[src1], dst1); -> bf16
    zero_f32_kernel<<<(int)((n4x + 255) / 256), 256, 0, stream>>>(agg1, n4x);
    long tot1 = (long)E1 * 128;
    scatter_add_f32_kernel<<<(int)((tot1 + 255) / 256), 256, 0, stream>>>(
        x, src1, dst1, agg1, tot1);
    cvt_f32_bf16_kernel<<<(int)((n4x + 255) / 256), 256, 0, stream>>>(agg1, agg1b, n4x);

    // 3) hb = relu(agg1b@Wrel1 + brel1 + xb@Wroot1)
    int mT1 = N1 / 16;                               // 1250
    int blocks1 = 8 * ((mT1 + 7) / 8);               // 8 n-groups x m-groups
    gemm_wmma_kernel<true, true, true><<<blocks1, 256, 0, stream>>>(
        agg1b, Wt_rel1, xb, Wt_root1, brel1, hb, mT1);

    // 4) Layer 2 aggregation from bf16 hb; -> bf16
    long n4b = (long)R2B / 4;
    zero_f32_kernel<<<(int)((n4b + 255) / 256), 256, 0, stream>>>(agg2, n4b);
    long tot2 = (long)E2 * 128;
    scatter_add_bf16_kernel<<<(int)((tot2 + 255) / 256), 256, 0, stream>>>(
        hb, src2, dst2, agg2, tot2);
    cvt_f32_bf16_kernel<<<(int)((n4b + 255) / 256), 256, 0, stream>>>(agg2, agg2b, n4b);

    // 5) h2b = agg2b@Wrel2 + brel2 + hb[:N2]@Wroot2
    int mT2 = N2 / 16;                               // 250
    int blocks2 = 8 * ((mT2 + 7) / 8);
    gemm_wmma_kernel<false, true, true><<<blocks2, 256, 0, stream>>>(
        agg2b, Wt_rel2, hb, Wt_root2, brel2, h2b, mT2);

    // 6) h3 = h2b@Wlin + blin   (f32 out for the head)
    gemm_wmma_kernel<false, false, false><<<blocks2, 256, 0, stream>>>(
        h2b, Wt_lin, nullptr, nullptr, blin, h3, mT2);

    // 7) out = h3@Whead + bhead  (4000x10)
    int totH = N2 * 10;
    head_kernel<<<(totH + 255) / 256, 256, 0, stream>>>(h3, Whead, bhead, out, totH);
}
